// MambaLayer_68745246539926
// MI455X (gfx1250) — matrix operand order repaired and checked by reading, not compiled, not run on hardware
//
#include <hip/hip_runtime.h>

typedef __attribute__((ext_vector_type(16))) _Float16 v16h;
typedef __attribute__((ext_vector_type(8)))  float    v8f;

#define NB    2
#define NSEQ  8000
#define NROWS (NB*NSEQ)      // 16000
#define DIM   96
#define DIN   192            // D_INNER
#define DST   16             // D_STATE
#define DTR   6              // DT_RANK
#define XDBL  38             // DTR + 2*DST
#define CHUNK 80
#define NG    100            // NSEQ / CHUNK

// ---------- WMMA fragment index helpers (ISA 7.12.2, wave32) ----------
__device__ __forceinline__ int a_k_of(int lane, int i) {
  int grp = lane >> 4;
  int v = i >> 1, h = i & 1;
  int k = (v < 4) ? (2*v + h) : (16 + 2*(v-4) + h);
  return k + 8*grp;              // lanes 0-15: K 0-7,16-23 ; lanes 16-31: K 8-15,24-31
}
__device__ __forceinline__ int b_k_of(int lane, int i) {
  return ((i < 8) ? i : (i + 8)) + 8*(lane >> 4);     // lanes 0-15: K 0-7,16-23 ; lanes 16-31: K 8-15,24-31
}

// CDNA5 async global->LDS DMA (GLOBAL_LOAD_ASYNC_TO_LDS_B128, tracked by ASYNCcnt).
// lds_off is relative to the block's LDS base; addr is a 64-bit global address.
__device__ __forceinline__ void async_copy_b128(unsigned lds_off, unsigned long long addr) {
  asm volatile("global_load_async_to_lds_b128 %0, %1, off"
               :: "v"(lds_off), "v"(addr) : "memory");
}
__device__ __forceinline__ void wait_asynccnt0() {
  asm volatile("s_wait_asynccnt 0x0" ::: "memory");
}

// ---------- prep: f32 -> f16 weight conversion ----------
__global__ void cvt_f16_kernel(const float* __restrict__ src, _Float16* __restrict__ dst, int n) {
  int i = blockIdx.x * 256 + threadIdx.x;
  if (i < n) dst[i] = (_Float16)src[i];
}

// ---------- 1. principal-axis projection ----------
__global__ void proj_kernel(const float* __restrict__ pv, float* __restrict__ proj) {
  int i = blockIdx.x * 256 + threadIdx.x;
  if (i >= NSEQ) return;
  int z = i / 400, y = (i / 20) % 20, x = i % 20;
  proj[i] = (float)z * pv[0] + (float)y * pv[1] + (float)x * pv[2];
}

// ---------- 2. stable argsort via O(N^2) ranking (L2-resident) ----------
__global__ void rank_kernel(const float* __restrict__ proj, int* __restrict__ sorted_idx) {
  int i = blockIdx.x * 256 + threadIdx.x;
  if (i >= NSEQ) return;
  float pi = proj[i];
  int cnt = 0;
  for (int j = 0; j < NSEQ; ++j) {
    float pj = proj[j];
    cnt += (pj < pi) || (pj == pi && j < i);
  }
  sorted_idx[cnt] = i;
}

// ---------- 3. gather + LayerNorm -> f16 activations ----------
__global__ void gather_ln_kernel(const float* __restrict__ x, const int* __restrict__ sorted_idx,
                                 const float* __restrict__ nw, const float* __restrict__ nb,
                                 _Float16* __restrict__ xn) {
  int row  = blockIdx.x * 8 + (threadIdx.x >> 5);   // one wave32 per row
  int lane = threadIdx.x & 31;
  int b = row / NSEQ, n = row % NSEQ;
  int pos = sorted_idx[n];
  float v[3], s = 0.f, ss = 0.f;
  #pragma unroll
  for (int j = 0; j < 3; ++j) {
    int c = lane + 32*j;
    v[j] = x[((size_t)b*DIM + c)*NSEQ + pos];
    s += v[j]; ss += v[j]*v[j];
  }
  #pragma unroll
  for (int m = 16; m >= 1; m >>= 1) { s += __shfl_xor(s, m, 32); ss += __shfl_xor(ss, m, 32); }
  float mu = s * (1.f/DIM);
  float rstd = rsqrtf(ss * (1.f/DIM) - mu*mu + 1e-5f);
  #pragma unroll
  for (int j = 0; j < 3; ++j) {
    int c = lane + 32*j;
    xn[(size_t)row*DIM + c] = (_Float16)(((v[j]-mu)*rstd)*nw[c] + nb[c]);
  }
}

// ---------- 4. in_proj GEMM: 16000x96 @ 96x384 via v_wmma_f32_16x16x32_f16 ----------
__global__ void gemm_in_kernel(const _Float16* __restrict__ xn, const _Float16* __restrict__ w16,
                               float* __restrict__ xz) {
  __shared__ _Float16 At[16][DIM];        // first LDS object -> block LDS offset 0
  int mtile = blockIdx.x;                 // 1000 tiles of 16 rows
  int t = threadIdx.x;                    // 256 = 8 waves
  // Stage the 16x96 f16 A-tile (3072 contiguous bytes) via async DMA to LDS.
  if (t < (16*DIM*2)/16) {                // 192 x b128 transfers
    unsigned long long ga = (unsigned long long)(uintptr_t)(xn + (size_t)mtile*16*DIM)
                          + (unsigned)(t*16);
    async_copy_b128((unsigned)(t*16), ga);
  }
  wait_asynccnt0();
  __syncthreads();
  int wave = t >> 5, lane = t & 31;
  int col = lane & 15, grp = lane >> 4;
  for (int nt = wave*3; nt < wave*3 + 3; ++nt) {   // 8 waves x 3 = 24 N-tiles (384 cols)
    v8f acc = {};
    #pragma unroll
    for (int kt = 0; kt < 3; ++kt) {               // K = 96 = 3 x 32
      __builtin_prefetch(&w16[(size_t)(nt*16 + col)*DIM + ((kt+1)%3)*32], 0, 0);
      v16h a, b;
      #pragma unroll
      for (int i = 0; i < 16; ++i) {
        a[i] = At[lane & 15][kt*32 + a_k_of(lane, i)];
        b[i] = w16[(size_t)(nt*16 + col)*DIM + kt*32 + b_k_of(lane, i)];
      }
      acc = __builtin_amdgcn_wmma_f32_16x16x32_f16(false, a, false, b, (short)0, acc, false, false);
    }
    #pragma unroll
    for (int r = 0; r < 8; ++r) {
      int row = mtile*16 + r + 8*grp;
      xz[(size_t)row*(2*DIN) + nt*16 + col] = acc[r];
    }
  }
}

// ---------- 5. depthwise causal conv(4) + SiLU ----------
__global__ void conv_silu_kernel(const float* __restrict__ xz, const float* __restrict__ cw,
                                 const float* __restrict__ cb, float* __restrict__ xc) {
  int idx = blockIdx.x * 256 + threadIdx.x;
  if (idx >= NROWS*DIN) return;
  int d = idx % DIN, row = idx / DIN;
  int n = row % NSEQ, b = row / NSEQ;
  float s = cb[d];
  #pragma unroll
  for (int j = 0; j < 4; ++j) {
    int nn = n - 3 + j;
    if (nn >= 0) s += cw[d*4 + j] * xz[(size_t)(b*NSEQ + nn)*(2*DIN) + d];
  }
  xc[(size_t)row*DIN + d] = s / (1.f + __expf(-s));
}

// ---------- 6. x_proj (38 cols, scalar dot) ----------
__global__ void xproj_kernel(const float* __restrict__ xc, const float* __restrict__ w,
                             float* __restrict__ xdbl) {
  int idx = blockIdx.x * 256 + threadIdx.x;
  if (idx >= NROWS*XDBL) return;
  int e = idx % XDBL, row = idx / XDBL;
  const float* xr = &xc[(size_t)row*DIN];
  const float* wr = &w[(size_t)e*DIN];
  float s = 0.f;
  for (int d = 0; d < DIN; ++d) s += xr[d] * wr[d];
  xdbl[(size_t)row*XDBL + e] = s;
}

// ---------- 7. dt_proj + softplus ----------
__global__ void dt_kernel(const float* __restrict__ xdbl, const float* __restrict__ w,
                          const float* __restrict__ bias, float* __restrict__ dt) {
  int idx = blockIdx.x * 256 + threadIdx.x;
  if (idx >= NROWS*DIN) return;
  int d = idx % DIN, row = idx / DIN;
  float s = bias[d];
  #pragma unroll
  for (int r = 0; r < DTR; ++r) s += xdbl[(size_t)row*XDBL + r] * w[d*DTR + r];
  dt[(size_t)row*DIN + d] = (s > 20.f) ? s : log1pf(__expf(s));
}

// ---------- 8a. chunked scan, phase 1: per-chunk carries ----------
__global__ void scan_p1_kernel(const float* __restrict__ dt, const float* __restrict__ xdbl,
                               const float* __restrict__ xc, const float* __restrict__ A_log,
                               float* __restrict__ aC, float* __restrict__ hC) {
  int g = blockIdx.x % NG, b = blockIdx.x / NG;
  int d = threadIdx.x;                       // 192 threads
  __shared__ float Bs[CHUNK][DST];
  for (int idx = d; idx < CHUNK*DST; idx += DIN) {
    int nn = idx / DST, s = idx % DST;
    Bs[nn][s] = xdbl[(size_t)(b*NSEQ + g*CHUNK + nn)*XDBL + DTR + s];
  }
  __syncthreads();
  float A[DST], h[DST];
  #pragma unroll
  for (int s = 0; s < DST; ++s) { A[s] = -__expf(A_log[d*DST + s]); h[s] = 0.f; }
  float dtsum = 0.f;
  for (int nn = 0; nn < CHUNK; ++nn) {
    size_t row = (size_t)(b*NSEQ + g*CHUNK + nn);
    float dtv = dt[row*DIN + d];
    float dx  = dtv * xc[row*DIN + d];
    dtsum += dtv;
    #pragma unroll
    for (int s = 0; s < DST; ++s) {
      float dA = __expf(dtv * A[s]);
      h[s] = dA * h[s] + dx * Bs[nn][s];
    }
  }
  size_t base = ((size_t)(b*NG + g)*DIN + d)*DST;
  #pragma unroll
  for (int s = 0; s < DST; ++s) {
    aC[base + s] = __expf(A[s] * dtsum);   // prod of exp(dt*A) over chunk, closed form
    hC[base + s] = h[s];
  }
}

// ---------- 8b. phase 2: serial combine over 100 chunk carries ----------
__global__ void scan_p2_kernel(const float* __restrict__ aC, const float* __restrict__ hC,
                               float* __restrict__ start) {
  int tid = blockIdx.x * 256 + threadIdx.x;
  if (tid >= NB*DIN*DST) return;
  int b = tid / (DIN*DST), ds = tid % (DIN*DST);
  float S = 0.f;
  for (int g = 0; g < NG; ++g) {
    size_t idx = (size_t)(b*NG + g)*(DIN*DST) + ds;
    start[idx] = S;
    S = aC[idx]*S + hC[idx];
  }
}

// ---------- 8c. phase 3: fix-up scan + y = (h.C + D*xc) * silu(z) -> f16 ----------
__global__ void scan_p3_kernel(const float* __restrict__ dt, const float* __restrict__ xdbl,
                               const float* __restrict__ xc, const float* __restrict__ xz,
                               const float* __restrict__ A_log, const float* __restrict__ Dp,
                               const float* __restrict__ start, _Float16* __restrict__ yh) {
  int g = blockIdx.x % NG, b = blockIdx.x / NG;
  int d = threadIdx.x;
  __shared__ float Bs[CHUNK][DST];
  __shared__ float Cs[CHUNK][DST];
  for (int idx = d; idx < CHUNK*DST; idx += DIN) {
    int nn = idx / DST, s = idx % DST;
    size_t row = (size_t)(b*NSEQ + g*CHUNK + nn);
    Bs[nn][s] = xdbl[row*XDBL + DTR + s];
    Cs[nn][s] = xdbl[row*XDBL + DTR + DST + s];
  }
  __syncthreads();
  float A[DST], h[DST];
  size_t sbase = ((size_t)(b*NG + g)*DIN + d)*DST;
  #pragma unroll
  for (int s = 0; s < DST; ++s) { A[s] = -__expf(A_log[d*DST + s]); h[s] = start[sbase + s]; }
  float Dv = Dp[d];
  for (int nn = 0; nn < CHUNK; ++nn) {
    size_t row = (size_t)(b*NSEQ + g*CHUNK + nn);
    float dtv = dt[row*DIN + d];
    float xcv = xc[row*DIN + d];
    float dx  = dtv * xcv;
    float y = 0.f;
    #pragma unroll
    for (int s = 0; s < DST; ++s) {
      float dA = __expf(dtv * A[s]);
      h[s] = dA * h[s] + dx * Bs[nn][s];
      y += h[s] * Cs[nn][s];
    }
    y += Dv * xcv;
    float zv = xz[row*(2*DIN) + DIN + d];
    y *= zv / (1.f + __expf(-zv));
    yh[row*DIN + d] = (_Float16)y;
  }
}

// ---------- 9. out_proj GEMM (16000x192 @ 192x96) + scatter via sorted_idx ----------
__global__ void gemm_out_kernel(const _Float16* __restrict__ yh, const _Float16* __restrict__ w16,
                                const int* __restrict__ sorted_idx, float* __restrict__ out) {
  __shared__ _Float16 At[16][DIN];        // first LDS object -> block LDS offset 0
  int mtile = blockIdx.x;                 // 1000 tiles
  int t = threadIdx.x;                    // 192 = 6 waves
  // Stage the 16x192 f16 A-tile (6144 contiguous bytes) via async DMA: 2 x b128 per thread.
  {
    unsigned long long gbase = (unsigned long long)(uintptr_t)(yh + (size_t)mtile*16*DIN);
    async_copy_b128((unsigned)(t*16),        gbase + (unsigned)(t*16));
    async_copy_b128((unsigned)((t+DIN)*16),  gbase + (unsigned)((t+DIN)*16));
  }
  wait_asynccnt0();
  __syncthreads();
  int wave = t >> 5, lane = t & 31;
  int col = lane & 15, grp = lane >> 4;
  int nt = wave;                          // 6 N-tiles (96 cols)
  v8f acc = {};
  #pragma unroll
  for (int kt = 0; kt < 6; ++kt) {        // K = 192 = 6 x 32
    __builtin_prefetch(&w16[(size_t)(nt*16 + col)*DIN + ((kt+1)%6)*32], 0, 0);
    v16h a, b;
    #pragma unroll
    for (int i = 0; i < 16; ++i) {
      a[i] = At[lane & 15][kt*32 + a_k_of(lane, i)];
      b[i] = w16[(size_t)(nt*16 + col)*DIN + kt*32 + b_k_of(lane, i)];
    }
    acc = __builtin_amdgcn_wmma_f32_16x16x32_f16(false, a, false, b, (short)0, acc, false, false);
  }
  #pragma unroll
  for (int r = 0; r < 8; ++r) {
    int grow = mtile*16 + r + 8*grp;
    int b2 = grow / NSEQ, n = grow % NSEQ;
    int pos = sorted_idx[n];
    out[((size_t)b2*DIM + nt*16 + col)*NSEQ + pos] = acc[r];
  }
}

// =====================================================================
extern "C" void kernel_launch(void* const* d_in, const int* in_sizes, int n_in,
                              void* d_out, int out_size, void* d_ws, size_t ws_size,
                              hipStream_t stream) {
  (void)in_sizes; (void)n_in; (void)out_size; (void)ws_size;
  const float* x        = (const float*)d_in[0];
  const float* pv       = (const float*)d_in[1];
  const float* norm_w   = (const float*)d_in[2];
  const float* norm_b   = (const float*)d_in[3];
  const float* in_w     = (const float*)d_in[4];
  const float* conv_w   = (const float*)d_in[5];
  const float* conv_b   = (const float*)d_in[6];
  const float* xproj_w  = (const float*)d_in[7];
  const float* dtp_w    = (const float*)d_in[8];
  const float* dtp_b    = (const float*)d_in[9];
  const float* A_log    = (const float*)d_in[10];
  const float* D_param  = (const float*)d_in[11];
  const float* out_w    = (const float*)d_in[12];
  float* out = (float*)d_out;

  char* ws = (char*)d_ws; size_t off = 0;
  auto alloc = [&](size_t bytes) { void* p = ws + off; off = (off + bytes + 255) & ~(size_t)255; return p; };
  _Float16* w16_in  = (_Float16*)alloc((size_t)2*DIN*DIM*sizeof(_Float16));
  _Float16* w16_out = (_Float16*)alloc((size_t)DIM*DIN*sizeof(_Float16));
  float*    proj    = (float*)alloc(NSEQ*sizeof(float));
  int*      sidx    = (int*)alloc(NSEQ*sizeof(int));
  _Float16* xn      = (_Float16*)alloc((size_t)NROWS*DIM*sizeof(_Float16));
  float*    xz      = (float*)alloc((size_t)NROWS*2*DIN*sizeof(float));
  float*    xc      = (float*)alloc((size_t)NROWS*DIN*sizeof(float));
  float*    xdbl    = (float*)alloc((size_t)NROWS*XDBL*sizeof(float));
  float*    dt      = (float*)alloc((size_t)NROWS*DIN*sizeof(float));
  float*    aC      = (float*)alloc((size_t)NB*NG*DIN*DST*sizeof(float));
  float*    hC      = (float*)alloc((size_t)NB*NG*DIN*DST*sizeof(float));
  float*    hstart  = (float*)alloc((size_t)NB*NG*DIN*DST*sizeof(float));
  _Float16* yh      = (_Float16*)alloc((size_t)NROWS*DIN*sizeof(_Float16));

  cvt_f16_kernel<<<(2*DIN*DIM + 255)/256, 256, 0, stream>>>(in_w, w16_in, 2*DIN*DIM);
  cvt_f16_kernel<<<(DIM*DIN + 255)/256, 256, 0, stream>>>(out_w, w16_out, DIM*DIN);
  proj_kernel<<<(NSEQ + 255)/256, 256, 0, stream>>>(pv, proj);
  rank_kernel<<<(NSEQ + 255)/256, 256, 0, stream>>>(proj, sidx);
  gather_ln_kernel<<<NROWS/8, 256, 0, stream>>>(x, sidx, norm_w, norm_b, xn);
  gemm_in_kernel<<<NROWS/16, 256, 0, stream>>>(xn, w16_in, xz);
  conv_silu_kernel<<<(NROWS*DIN + 255)/256, 256, 0, stream>>>(xz, conv_w, conv_b, xc);
  xproj_kernel<<<(NROWS*XDBL + 255)/256, 256, 0, stream>>>(xc, xproj_w, xdbl);
  dt_kernel<<<(NROWS*DIN + 255)/256, 256, 0, stream>>>(xdbl, dtp_w, dtp_b, dt);
  scan_p1_kernel<<<NB*NG, DIN, 0, stream>>>(dt, xdbl, xc, A_log, aC, hC);
  scan_p2_kernel<<<(NB*DIN*DST + 255)/256, 256, 0, stream>>>(aC, hC, hstart);
  scan_p3_kernel<<<NB*NG, DIN, 0, stream>>>(dt, xdbl, xc, xz, A_log, D_param, hstart, yh);
  gemm_out_kernel<<<NROWS/16, DIN, 0, stream>>>(yh, w16_out, sidx, out);
}
